// MiMoV2Attention_70196945486461
// MI455X (gfx1250) — compile-verified, matrix operand order checked by reading, not run on hardware
//
#include <hip/hip_runtime.h>
#include <hip/hip_bf16.h>
#include <stdint.h>

typedef __attribute__((ext_vector_type(16))) __bf16 v16bf;
typedef __attribute__((ext_vector_type(2)))  __bf16 v2bf;
typedef __attribute__((ext_vector_type(8)))  float  v8f;

#define HIDDEN 4096
#define NQ 32
#define NKV 8
#define HD 128
#define WINDOW 1024
#define SEQ 2048

// fp32 pair -> packed bf16 dword via hardware converter (v_cvt_pk_bf16_f32)
__device__ __forceinline__ unsigned int pack_bf(float lo, float hi) {
#if __has_builtin(__builtin_amdgcn_cvt_pk_bf16_f32)
  v2bf p = __builtin_amdgcn_cvt_pk_bf16_f32(lo, hi);
  return __builtin_bit_cast(unsigned int, p);
#else
  v2bf p;
  p.x = (__bf16)lo;   // v_cvt_bf16_f32 into low half
  p.y = (__bf16)hi;   // v_cvt_bf16_f32 into high half
  return __builtin_bit_cast(unsigned int, p);
#endif
}
__device__ __forceinline__ unsigned short f2bf1(float f) {
  __bf16 b = (__bf16)f;
  return __builtin_bit_cast(unsigned short, b);
}

union FragB16 { v16bf v; unsigned int u[8]; };
union AccU    { v8f   v; float f[8]; };

// ---------------------------------------------------------------------------
// C[M,N] = A[M,K] (fp32, row-major) * B[K,N] (fp32, row-major), bf16 WMMA,
// fp32 accumulation. N,K compile-time so all strides fold into immediate
// offsets. Block tile 128x128, K-step 32, 256 threads = 8 waves (wave32),
// wave tile 32x64. Double-buffered LDS + L2 prefetch two tiles ahead.
// ---------------------------------------------------------------------------
template <int N, int K>
__global__ __launch_bounds__(256) void gemm_bf16(
    const float* __restrict__ A, const float* __restrict__ B,
    float* __restrict__ C) {
  __shared__ unsigned int lds_a[2][128 * 16];  // [buf][row][k-pair]
  __shared__ unsigned int lds_b[2][128 * 16];  // [buf][col][k-pair] (B^T)

  const int tid  = threadIdx.x;
  const int lane = tid & 31;
  const int wid  = tid >> 5;
  const int half = lane >> 4;
  const int idx  = lane & 15;
  const int waveM = wid & 3;
  const int waveN = wid >> 2;
  const int m0 = blockIdx.y * 128;
  const int n0 = blockIdx.x * 128;

  const int arow = tid >> 1;        // 0..127
  const int ach  = tid & 1;         // 16-col half of A tile
  const int bcol = tid & 127;       // column of B tile
  const int bkh  = tid >> 7;        // 16-row half of B tile (K dim)

  const float* aptr = A + (size_t)(m0 + arow) * K + ach * 16;
  const float* bptr = B + (size_t)(bkh * 16) * N + n0 + bcol;

  AccU acc[2][4];
#pragma unroll
  for (int i = 0; i < 2; i++)
#pragma unroll
    for (int j = 0; j < 4; j++)
#pragma unroll
      for (int e = 0; e < 8; e++) acc[i][j].f[e] = 0.0f;

  float4 fa0, fa1, fa2, fa3;
  float  fb[16];

  // preload tile 0 into registers, then LDS buffer 0
  {
    const float4* p = (const float4*)aptr;
    fa0 = p[0]; fa1 = p[1]; fa2 = p[2]; fa3 = p[3];
#pragma unroll
    for (int j = 0; j < 16; j++) fb[j] = bptr[(size_t)j * N];
  }
  {
    unsigned int* da = &lds_a[0][arow * 16 + ach * 8];
    da[0] = pack_bf(fa0.x, fa0.y); da[1] = pack_bf(fa0.z, fa0.w);
    da[2] = pack_bf(fa1.x, fa1.y); da[3] = pack_bf(fa1.z, fa1.w);
    da[4] = pack_bf(fa2.x, fa2.y); da[5] = pack_bf(fa2.z, fa2.w);
    da[6] = pack_bf(fa3.x, fa3.y); da[7] = pack_bf(fa3.z, fa3.w);
    unsigned int* db = &lds_b[0][bcol * 16 + bkh * 8];
#pragma unroll
    for (int j = 0; j < 8; j++) db[j] = pack_bf(fb[2 * j], fb[2 * j + 1]);
  }

  int buf = 0;
  for (int k0 = 0; k0 < K; k0 += 32) {
    __syncthreads();
    const bool nxt = (k0 + 32) < K;
    if (nxt) {
      const float4* p = (const float4*)(aptr + k0 + 32);
      fa0 = p[0]; fa1 = p[1]; fa2 = p[2]; fa3 = p[3];
#pragma unroll
      for (int j = 0; j < 16; j++)
        fb[j] = bptr[(size_t)(k0 + 32 + j) * N];
      if (k0 + 64 < K) {  // pull tile k+2 toward L2 (global_prefetch_b8)
        __builtin_prefetch(aptr + k0 + 64, 0, 1);
        __builtin_prefetch(bptr + (size_t)(k0 + 64) * N, 0, 1);
      }
    }

    FragB16 af[2], bf[4];
#pragma unroll
    for (int mt = 0; mt < 2; mt++) {
      int r = waveM * 32 + mt * 16 + idx;
      const unsigned int* p = &lds_a[buf][r * 16];
#pragma unroll
      for (int j = 0; j < 4; j++) {
        af[mt].u[j]     = p[half * 4 + j];        // K = half*8 + 0..7
        af[mt].u[4 + j] = p[8 + half * 4 + j];    // K = 16 + half*8 + 0..7
      }
    }
#pragma unroll
    for (int nt = 0; nt < 4; nt++) {
      int c = waveN * 64 + nt * 16 + idx;
      const unsigned int* p = &lds_b[buf][c * 16];
#pragma unroll
      for (int j = 0; j < 4; j++) {
        bf[nt].u[j]     = p[half * 4 + j];
        bf[nt].u[4 + j] = p[8 + half * 4 + j];
      }
    }
#pragma unroll
    for (int mt = 0; mt < 2; mt++)
#pragma unroll
      for (int nt = 0; nt < 4; nt++)
        acc[mt][nt].v = __builtin_amdgcn_wmma_f32_16x16x32_bf16(
            false, af[mt].v, false, bf[nt].v, (short)0, acc[mt][nt].v,
            false, false);

    if (nxt) {  // stage tile k+1 into the other buffer (no barrier needed)
      unsigned int* da = &lds_a[buf ^ 1][arow * 16 + ach * 8];
      da[0] = pack_bf(fa0.x, fa0.y); da[1] = pack_bf(fa0.z, fa0.w);
      da[2] = pack_bf(fa1.x, fa1.y); da[3] = pack_bf(fa1.z, fa1.w);
      da[4] = pack_bf(fa2.x, fa2.y); da[5] = pack_bf(fa2.z, fa2.w);
      da[6] = pack_bf(fa3.x, fa3.y); da[7] = pack_bf(fa3.z, fa3.w);
      unsigned int* db = &lds_b[buf ^ 1][bcol * 16 + bkh * 8];
#pragma unroll
      for (int j = 0; j < 8; j++) db[j] = pack_bf(fb[2 * j], fb[2 * j + 1]);
      buf ^= 1;
    }
  }

  // epilogue: one base per lane, all offsets are compile-time immediates
#pragma unroll
  for (int mt = 0; mt < 2; mt++) {
    float* cp = C + (size_t)(m0 + waveM * 32 + mt * 16 + half * 8) * N + n0 +
                waveN * 64 + idx;
#pragma unroll
    for (int nt = 0; nt < 4; nt++)
#pragma unroll
      for (int r = 0; r < 8; r++)
        cp[(size_t)r * N + nt * 16] = acc[mt][nt].f[r];
  }
}

// ---------------------------------------------------------------------------
// NeoX RoPE in-place on [SEQ, n_heads, 128]
// ---------------------------------------------------------------------------
__global__ void rope_kernel(float* __restrict__ x,
                            const int* __restrict__ positions, int n_heads) {
  int t = blockIdx.x * blockDim.x + threadIdx.x;
  int total = SEQ * n_heads * (HD / 2);
  if (t >= total) return;
  int dp = t & 63;
  int h  = (t >> 6) % n_heads;   // n_heads is a power of two here
  int s  = t / (n_heads * 64);
  float pos = (float)positions[s];
  float inv = __expf(-(float)dp * (13.815510557964274f / 64.0f));  // theta^-dp/64
  float fr = pos * inv;
  float c = __cosf(fr), sn = __sinf(fr);
  float* p = x + (size_t)s * n_heads * HD + (size_t)h * HD + dp;
  float x1 = p[0], x2 = p[64];
  p[0]  = x1 * c - x2 * sn;
  p[64] = x2 * c + x1 * sn;
}

// ---------------------------------------------------------------------------
// Flash attention with sliding window + sink; one wave per (head, 16 queries)
// ---------------------------------------------------------------------------
__device__ __forceinline__ v8f qk_tile(const FragB16* qf, const float* kp,
                                       int half) {
  AccU s;
#pragma unroll
  for (int e = 0; e < 8; e++) s.f[e] = 0.0f;
#pragma unroll
  for (int st = 0; st < 4; st++) {
    FragB16 kf;
    int b0 = st * 32 + half * 8;
    float4 f0 = ((const float4*)(kp + b0))[0];
    float4 f1 = ((const float4*)(kp + b0))[1];
    float4 f2 = ((const float4*)(kp + b0 + 16))[0];
    float4 f3 = ((const float4*)(kp + b0 + 16))[1];
    kf.u[0] = pack_bf(f0.x, f0.y); kf.u[1] = pack_bf(f0.z, f0.w);
    kf.u[2] = pack_bf(f1.x, f1.y); kf.u[3] = pack_bf(f1.z, f1.w);
    kf.u[4] = pack_bf(f2.x, f2.y); kf.u[5] = pack_bf(f2.z, f2.w);
    kf.u[6] = pack_bf(f3.x, f3.y); kf.u[7] = pack_bf(f3.z, f3.w);
    s.v = __builtin_amdgcn_wmma_f32_16x16x32_bf16(
        false, qf[st].v, false, kf.v, (short)0, s.v, false, false);
  }
  return s.v;
}

__global__ __launch_bounds__(32) void attn_kernel(
    const float* __restrict__ q, const float* __restrict__ k,
    const float* __restrict__ v, const float* __restrict__ sink,
    float* __restrict__ out) {
  __shared__ unsigned int lds_p[16 * 16];
  unsigned short* lds_ps = (unsigned short*)lds_p;

  const int lane = threadIdx.x;
  const int half = lane >> 4, idx = lane & 15;
  const int qb = blockIdx.x;
  const int h  = blockIdx.y;
  const int hkv = h >> 2;          // 4 q heads per kv head (GQA)
  const int q0 = qb * 16;
  const float scaling = 0.08838834764831845f;  // 128^-0.5

  FragB16 qf[4];
  {
    const float* qp = q + (size_t)(q0 + idx) * (NQ * HD) + (size_t)h * HD;
#pragma unroll
    for (int st = 0; st < 4; st++) {
      int b0 = st * 32 + half * 8;
      float4 f0 = ((const float4*)(qp + b0))[0];
      float4 f1 = ((const float4*)(qp + b0))[1];
      float4 f2 = ((const float4*)(qp + b0 + 16))[0];
      float4 f3 = ((const float4*)(qp + b0 + 16))[1];
      qf[st].u[0] = pack_bf(f0.x, f0.y); qf[st].u[1] = pack_bf(f0.z, f0.w);
      qf[st].u[2] = pack_bf(f1.x, f1.y); qf[st].u[3] = pack_bf(f1.z, f1.w);
      qf[st].u[4] = pack_bf(f2.x, f2.y); qf[st].u[5] = pack_bf(f2.z, f2.w);
      qf[st].u[6] = pack_bf(f3.x, f3.y); qf[st].u[7] = pack_bf(f3.z, f3.w);
    }
  }

  float mrow[8], lrow[8];
  AccU acc[8];
  float sk = sink[h];
#pragma unroll
  for (int r = 0; r < 8; r++) { mrow[r] = sk; lrow[r] = 1.0f; }  // sink term
#pragma unroll
  for (int nt = 0; nt < 8; nt++)
#pragma unroll
    for (int e = 0; e < 8; e++) acc[nt].f[e] = 0.0f;

  int j_lo = q0 - (WINDOW - 1); if (j_lo < 0) j_lo = 0;
  const int kb_lo = j_lo >> 5;
  const int kb_hi = (q0 + 15) >> 5;

  for (int kb = kb_lo; kb <= kb_hi; kb++) {
    AccU s0, s1;
    {
      const float* kp0 =
          k + (size_t)(kb * 32 + idx) * (NKV * HD) + (size_t)hkv * HD;
      const float* kp1 =
          k + (size_t)(kb * 32 + 16 + idx) * (NKV * HD) + (size_t)hkv * HD;
      s0.v = qk_tile(qf, kp0, half);
      s1.v = qk_tile(qf, kp1, half);
    }

    float p0v[8], p1v[8], alpha[8];
    int j0 = kb * 32 + idx;
    int j1 = kb * 32 + 16 + idx;
#pragma unroll
    for (int r = 0; r < 8; r++) {
      int i = q0 + r + half * 8;
      float a = (j0 <= i && (i - j0) < WINDOW) ? s0.f[r] * scaling
                                               : -__builtin_inff();
      float b = (j1 <= i && (i - j1) < WINDOW) ? s1.f[r] * scaling
                                               : -__builtin_inff();
      float tm = fmaxf(a, b);
      tm = fmaxf(tm, __shfl_xor(tm, 1, 32));
      tm = fmaxf(tm, __shfl_xor(tm, 2, 32));
      tm = fmaxf(tm, __shfl_xor(tm, 4, 32));
      tm = fmaxf(tm, __shfl_xor(tm, 8, 32));
      float mnew = fmaxf(mrow[r], tm);
      alpha[r] = __expf(mrow[r] - mnew);
      mrow[r] = mnew;
      float pa = __expf(a - mnew);
      float pb = __expf(b - mnew);
      p0v[r] = pa; p1v[r] = pb;
      float rs = pa + pb;
      rs += __shfl_xor(rs, 1, 32);
      rs += __shfl_xor(rs, 2, 32);
      rs += __shfl_xor(rs, 4, 32);
      rs += __shfl_xor(rs, 8, 32);
      lrow[r] = lrow[r] * alpha[r] + rs;
    }
#pragma unroll
    for (int nt = 0; nt < 8; nt++)
#pragma unroll
      for (int r = 0; r < 8; r++) acc[nt].f[r] *= alpha[r];

    // stage P (C-layout) through LDS into A-fragment layout, as bf16
#pragma unroll
    for (int r = 0; r < 8; r++) {
      int row = r + half * 8;
      lds_ps[row * 32 + idx]      = f2bf1(p0v[r]);
      lds_ps[row * 32 + 16 + idx] = f2bf1(p1v[r]);
    }
    __syncthreads();
    FragB16 pf;
    {
      const unsigned int* pr = lds_p + idx * 16;
#pragma unroll
      for (int j = 0; j < 4; j++) {
        pf.u[j]     = pr[half * 4 + j];
        pf.u[4 + j] = pr[8 + half * 4 + j];
      }
    }
    __syncthreads();

    // acc += P (16x32) * V (32x128), 8 N-tiles of 16 dims
#pragma unroll
    for (int nt = 0; nt < 8; nt++) {
      FragB16 vf;
      int d = nt * 16 + idx;
      const float* vp =
          v + (size_t)(kb * 32) * (NKV * HD) + (size_t)hkv * HD + d;
#pragma unroll
      for (int j = 0; j < 4; j++) {
        int ko = half * 8 + 2 * j;
        vf.u[j] = pack_bf(vp[(size_t)ko * (NKV * HD)],
                          vp[(size_t)(ko + 1) * (NKV * HD)]);
        int ko2 = 16 + ko;
        vf.u[4 + j] = pack_bf(vp[(size_t)ko2 * (NKV * HD)],
                              vp[(size_t)(ko2 + 1) * (NKV * HD)]);
      }
      acc[nt].v = __builtin_amdgcn_wmma_f32_16x16x32_bf16(
          false, pf.v, false, vf.v, (short)0, acc[nt].v, false, false);
    }
  }

  float invl[8];
#pragma unroll
  for (int r = 0; r < 8; r++) invl[r] = 1.0f / lrow[r];
#pragma unroll
  for (int nt = 0; nt < 8; nt++)
#pragma unroll
    for (int r = 0; r < 8; r++) {
      int row = q0 + r + half * 8;
      out[(size_t)row * (NQ * HD) + (size_t)h * HD + nt * 16 + idx] =
          acc[nt].f[r] * invl[r];
    }
}

extern "C" void kernel_launch(void* const* d_in, const int* in_sizes, int n_in,
                              void* d_out, int out_size, void* d_ws,
                              size_t ws_size, hipStream_t stream) {
  const float* hidden    = (const float*)d_in[0];
  const int*   positions = (const int*)d_in[1];
  const float* wq        = (const float*)d_in[2];
  const float* wk        = (const float*)d_in[3];
  const float* wv        = (const float*)d_in[4];
  const float* wo        = (const float*)d_in[5];
  const float* sink      = (const float*)d_in[6];
  float* out = (float*)d_out;

  float* qbuf = (float*)d_ws;                   // 32 MB
  float* kbuf = qbuf + (size_t)SEQ * NQ * HD;   //  8 MB
  float* vbuf = kbuf + (size_t)SEQ * NKV * HD;  //  8 MB

  gemm_bf16<NQ * HD, HIDDEN>
      <<<dim3((NQ * HD) / 128, SEQ / 128), 256, 0, stream>>>(hidden, wq, qbuf);
  gemm_bf16<NKV * HD, HIDDEN>
      <<<dim3((NKV * HD) / 128, SEQ / 128), 256, 0, stream>>>(hidden, wk, kbuf);
  gemm_bf16<NKV * HD, HIDDEN>
      <<<dim3((NKV * HD) / 128, SEQ / 128), 256, 0, stream>>>(hidden, wv, vbuf);

  int nrq = SEQ * NQ * (HD / 2);
  rope_kernel<<<(nrq + 255) / 256, 256, 0, stream>>>(qbuf, positions, NQ);
  int nrk = SEQ * NKV * (HD / 2);
  rope_kernel<<<(nrk + 255) / 256, 256, 0, stream>>>(kbuf, positions, NKV);

  // writes attention output in-place over qbuf (each wave owns its head slice)
  attn_kernel<<<dim3(SEQ / 16, NQ), 32, 0, stream>>>(qbuf, kbuf, vbuf, sink,
                                                     qbuf);

  gemm_bf16<HIDDEN, HIDDEN>
      <<<dim3(HIDDEN / 128, SEQ / 128), 256, 0, stream>>>(qbuf, wo, out);
}
